// GCN_39393440039563
// MI455X (gfx1250) — matrix operand order repaired and checked by reading, not compiled, not run on hardware
//
#include <hip/hip_runtime.h>

// ---------------------------------------------------------------------------
// GCN 2-layer forward for MI455X (gfx1250, wave32).
// GEMMs: v_wmma_f32_16x16x32_f16 with split-f16 (hi/lo) operands for
// near-fp32 accuracy (3 WMMAs per 16x16x32 tile), double-buffered LDS
// (1 barrier / k-tile), packed b64 LDS staging stores.
// Edge aggregation: L2-resident atomic f32 scatter (h matrices fit 192MB L2).
// ---------------------------------------------------------------------------

typedef __attribute__((ext_vector_type(16))) _Float16 v16h;
typedef __attribute__((ext_vector_type(8)))  _Float16 v8h;
typedef __attribute__((ext_vector_type(8)))  float    v8f;

#define BM 128
#define BN 64
#define BK 32
#define THREADS 256

__device__ __forceinline__ v16h ld_frag(const _Float16* p0, const _Float16* p1) {
  union { v16h v; v8h h[2]; } u;
  u.h[0] = *(const v8h*)p0;   // ds_load_b128
  u.h[1] = *(const v8h*)p1;   // ds_load_b128
  return u.v;
}

// split 4 f32 into packed 4xf16 hi and 4xf16 lo (two uint2 -> ds_store_b64)
__device__ __forceinline__ void split4(const float* f, uint2& hiP, uint2& loP) {
  union { _Float16 h[4]; uint2 u; } uh, ul;
#pragma unroll
  for (int j = 0; j < 4; ++j) {
    _Float16 hi = (_Float16)f[j];
    uh.h[j] = hi;
    ul.h[j] = (_Float16)(f[j] - (float)hi);
  }
  hiP = uh.u; loP = ul.u;
}

struct TileRegs {
  float4 a[4];   // A tile: 4 x float4 per thread (128x32 block)
  float  b[8];   // B tile: 8 scalars per thread (32x64 block, 4 consecutive K)
};

__device__ __forceinline__ void load_tile(const float* __restrict__ A,
                                          const float* __restrict__ B,
                                          int M, int N, int K,
                                          int m0, int n0, int k0,
                                          int tid, TileRegs& r) {
#pragma unroll
  for (int i = 0; i < 4; ++i) {              // A: 1024 float4 chunks / 256 thr
    int lin = tid + i * THREADS;
    int row = lin >> 3;
    int c   = (lin & 7) << 2;
    r.a[i] = make_float4(0.f, 0.f, 0.f, 0.f);
    if (m0 + row < M)
      r.a[i] = *(const float4*)(A + (size_t)(m0 + row) * K + k0 + c);
  }
#pragma unroll
  for (int i = 0; i < 2; ++i) {              // B: each thread 1 col x 4 K rows
    int lin = tid + i * THREADS;             // 0..511
    int n  = lin & 63;
    int kq = lin >> 6;                       // 0..7 (K quad)
#pragma unroll
    for (int j = 0; j < 4; ++j)              // coalesced across lanes (n varies)
      r.b[i * 4 + j] = B[(size_t)(k0 + kq * 4 + j) * N + n0 + n];
  }
}

__device__ __forceinline__ void store_tile(_Float16* sAhi, _Float16* sAlo,
                                           _Float16* sBhi, _Float16* sBlo,
                                           int tid, const TileRegs& r) {
#pragma unroll
  for (int i = 0; i < 4; ++i) {
    int lin = tid + i * THREADS;
    int row = lin >> 3;
    int c   = (lin & 7) << 2;
    float f[4] = {r.a[i].x, r.a[i].y, r.a[i].z, r.a[i].w};
    uint2 hp, lp; split4(f, hp, lp);
    *(uint2*)(sAhi + row * BK + c) = hp;     // ds_store_b64
    *(uint2*)(sAlo + row * BK + c) = lp;
  }
#pragma unroll
  for (int i = 0; i < 2; ++i) {
    int lin = tid + i * THREADS;
    int n  = lin & 63;
    int kq = lin >> 6;
    uint2 hp, lp; split4(&r.b[i * 4], hp, lp);
    *(uint2*)(sBhi + n * BK + kq * 4) = hp;  // ds_store_b64 ([n][k] transposed)
    *(uint2*)(sBlo + n * BK + kq * 4) = lp;
  }
}

// C[M,N] = A[M,K] * B[K,N], all f32 row-major. K % 32 == 0, N % 64 == 0.
__global__ __launch_bounds__(THREADS)
void gcn_gemm_wmma(const float* __restrict__ A, const float* __restrict__ B,
                   float* __restrict__ C, int M, int N, int K) {
  __shared__ __align__(16) _Float16 sAhi[2][BM * BK];
  __shared__ __align__(16) _Float16 sAlo[2][BM * BK];
  __shared__ __align__(16) _Float16 sBhi[2][BN * BK];  // [n][k] transposed
  __shared__ __align__(16) _Float16 sBlo[2][BN * BK];

  const int tid  = threadIdx.x;
  const int wid  = tid >> 5;        // 0..7 (8 waves, wave32)
  const int lane = tid & 31;
  const int half = lane >> 4;       // 0/1
  const int l16  = lane & 15;
  const int m0 = blockIdx.x * BM;
  const int n0 = blockIdx.y * BN;

  v8f acc[BN / 16] = {};            // 4 accumulators of 16x16 f32

  const int kTiles = K / BK;
  TileRegs r;

  // prologue: stage tile 0 into buffer 0
  load_tile(A, B, M, N, K, m0, n0, 0, tid, r);
  store_tile(sAhi[0], sAlo[0], sBhi[0], sBlo[0], tid, r);
  __syncthreads();

  for (int kt = 0; kt < kTiles; ++kt) {
    const int cur = kt & 1;
    const bool more = (kt + 1) < kTiles;

    // prefetch next tile into registers (overlaps with WMMA below)
    if (more) load_tile(A, B, M, N, K, m0, n0, (kt + 1) * BK, tid, r);

    // ---- compute from buffer `cur`: 1 A-frag reused over 4 N-subtiles ----
    const int arow = wid * 16 + l16;
    // A 16x32 f16 layout: elems 0..7 -> K = half*8+e ; 8..15 -> 16+half*8+e
    v16h ahi = ld_frag(&sAhi[cur][arow * BK + half * 8],
                       &sAhi[cur][arow * BK + 16 + half * 8]);
    v16h alo = ld_frag(&sAlo[cur][arow * BK + half * 8],
                       &sAlo[cur][arow * BK + 16 + half * 8]);
#pragma unroll
    for (int nt = 0; nt < BN / 16; ++nt) {
      const int bcol = nt * 16 + l16;
      // B 32x16 f16 layout: elems 0..7 -> K = half*8+e ; 8..15 -> 16+half*8+e (LDS [n][k])
      v16h bhi = ld_frag(&sBhi[cur][bcol * BK + half * 8],
                         &sBhi[cur][bcol * BK + 16 + half * 8]);
      v16h blo = ld_frag(&sBlo[cur][bcol * BK + half * 8],
                         &sBlo[cur][bcol * BK + 16 + half * 8]);
      acc[nt] = __builtin_amdgcn_wmma_f32_16x16x32_f16(false, ahi, false, bhi,
                                                       (short)0, acc[nt], false, false);
      acc[nt] = __builtin_amdgcn_wmma_f32_16x16x32_f16(false, alo, false, bhi,
                                                       (short)0, acc[nt], false, false);
      acc[nt] = __builtin_amdgcn_wmma_f32_16x16x32_f16(false, ahi, false, blo,
                                                       (short)0, acc[nt], false, false);
    }

    // stage next tile into the other buffer (no reader conflicts), 1 barrier
    if (more)
      store_tile(sAhi[1 - cur], sAlo[1 - cur], sBhi[1 - cur], sBlo[1 - cur], tid, r);
    __syncthreads();
  }

  // ---- store C: VGPR r -> row r + 8*half, col l16 ----------------------
#pragma unroll
  for (int nt = 0; nt < BN / 16; ++nt) {
    const int col = n0 + nt * 16 + l16;
#pragma unroll
    for (int rr = 0; rr < 8; ++rr) {
      const int row = m0 + wid * 16 + half * 8 + rr;
      if (row < M) C[(size_t)row * N + col] = acc[nt][rr];
    }
  }
}

// ---------------------------------------------------------------------------
// Degree / normalization
// ---------------------------------------------------------------------------
__global__ void init_deg(float* deg, int n) {
  int i = blockIdx.x * 256 + threadIdx.x;
  if (i < n) deg[i] = 1.0f;                  // self-loop folded in
}
__global__ void deg_edges(const int* __restrict__ dst, float* deg, int E) {
  int e = blockIdx.x * 256 + threadIdx.x;
  if (e < E) atomicAdd(&deg[dst[e]], 1.0f);
}
__global__ void make_dinv(float* deg, int n) {
  int i = blockIdx.x * 256 + threadIdx.x;
  if (i < n) deg[i] = rsqrtf(deg[i]);        // deg >= 1 always
}

// out[i,:] = h[i,:] * dinv[i]^2 (+ bias)   -- self-loop contribution & init
__global__ void agg_self(const float* __restrict__ h, const float* __restrict__ dinv,
                         const float* __restrict__ bias, float* __restrict__ out,
                         int n, int F, int fshift, int addBias) {
  size_t idx = (size_t)blockIdx.x * 256 + threadIdx.x;
  size_t total = (size_t)n << fshift;        // n * F/4
  if (idx >= total) return;
  int i = (int)(idx >> fshift);
  int c = (int)(idx & (((size_t)1 << fshift) - 1)) << 2;
  float w = dinv[i]; w = w * w;
  float4 v = *(const float4*)(h + (size_t)i * F + c);
  float4 o;
  o.x = v.x * w; o.y = v.y * w; o.z = v.z * w; o.w = v.w * w;
  if (addBias) { o.x += bias[c]; o.y += bias[c + 1]; o.z += bias[c + 2]; o.w += bias[c + 3]; }
  *(float4*)(out + (size_t)i * F + c) = o;
}

// out[dst,:] += h[src,:] * dinv[src]*dinv[dst]   (L2-resident scatter)
__global__ void agg_edges(const int* __restrict__ src, const int* __restrict__ dst,
                          const float* __restrict__ dinv, const float* __restrict__ h,
                          float* __restrict__ out, int E, int F, int cshift) {
  size_t idx = (size_t)blockIdx.x * 256 + threadIdx.x;
  size_t total = (size_t)E << cshift;        // E * F/4
  if (idx >= total) return;
  int e = (int)(idx >> cshift);
  int c = (int)(idx & (((size_t)1 << cshift) - 1)) << 2;
  int s = src[e], d = dst[e];
  float w = dinv[s] * dinv[d];
  float4 v = *(const float4*)(h + (size_t)s * F + c);
  float* o = out + (size_t)d * F + c;
  atomicAdd(o + 0, v.x * w);
  atomicAdd(o + 1, v.y * w);
  atomicAdd(o + 2, v.z * w);
  atomicAdd(o + 3, v.w * w);
}

// out[i,f] = relu(in[i,f] + bias[f])
__global__ void bias_relu(const float* __restrict__ in, const float* __restrict__ bias,
                          float* __restrict__ out, int n, int F, int fshift) {
  size_t idx = (size_t)blockIdx.x * 256 + threadIdx.x;
  size_t total = (size_t)n << fshift;
  if (idx >= total) return;
  int i = (int)(idx >> fshift);
  int c = (int)(idx & (((size_t)1 << fshift) - 1)) << 2;
  float4 v = *(const float4*)(in + (size_t)i * F + c);
  v.x = fmaxf(v.x + bias[c + 0], 0.f);
  v.y = fmaxf(v.y + bias[c + 1], 0.f);
  v.z = fmaxf(v.z + bias[c + 2], 0.f);
  v.w = fmaxf(v.w + bias[c + 3], 0.f);
  *(float4*)(out + (size_t)i * F + c) = v;
}

// ---------------------------------------------------------------------------
extern "C" void kernel_launch(void* const* d_in, const int* in_sizes, int n_in,
                              void* d_out, int out_size, void* d_ws, size_t ws_size,
                              hipStream_t stream) {
  const float* x  = (const float*)d_in[0];
  const int*   ei = (const int*)d_in[1];
  const float* W1 = (const float*)d_in[2];
  const float* b1 = (const float*)d_in[3];
  const float* W2 = (const float*)d_in[4];
  const float* b2 = (const float*)d_in[5];
  float* out = (float*)d_out;

  const int IN_F = 768, HID = 256, OUT_F = 128;
  const int n = in_sizes[0] / IN_F;
  const int E = in_sizes[1] / 2;
  const int* src = ei;          // edge_index[0]
  const int* dst = ei + E;      // edge_index[1]

  // workspace layout
  char* ws = (char*)d_ws;
  size_t off = 0;
  float* dinv = (float*)(ws + off); off += ((size_t)n * 4 + 255) & ~(size_t)255;
  float* h0   = (float*)(ws + off); off += ((size_t)n * HID * 4 + 255) & ~(size_t)255;
  float* agg1 = (float*)(ws + off); off += ((size_t)n * HID * 4 + 255) & ~(size_t)255;
  float* h2   = (float*)(ws + off);

  const int B = 256;
  // normalization: deg (with self-loop) -> dinv
  init_deg <<<(n + B - 1) / B, B, 0, stream>>>(dinv, n);
  deg_edges<<<(E + B - 1) / B, B, 0, stream>>>(dst, dinv, E);
  make_dinv<<<(n + B - 1) / B, B, 0, stream>>>(dinv, n);

  // layer 1: h0 = x @ W1
  dim3 g1((n + BM - 1) / BM, HID / BN);
  gcn_gemm_wmma<<<g1, THREADS, 0, stream>>>(x, W1, h0, n, HID, IN_F);

  size_t t1  = (size_t)n * (HID / 4);
  size_t te1 = (size_t)E * (HID / 4);
  agg_self <<<(t1 + B - 1) / B, B, 0, stream>>>(h0, dinv, nullptr, agg1, n, HID, 6, 0);
  agg_edges<<<(te1 + B - 1) / B, B, 0, stream>>>(src, dst, dinv, h0, agg1, E, HID, 6);
  bias_relu<<<(t1 + B - 1) / B, B, 0, stream>>>(agg1, b1, h0, n, HID, 6);  // h = relu(agg+b1)

  // layer 2: h2 = h @ W2
  dim3 g2((n + BM - 1) / BM, OUT_F / BN);
  gcn_gemm_wmma<<<g2, THREADS, 0, stream>>>(h0, W2, h2, n, OUT_F, HID);

  size_t t2  = (size_t)n * (OUT_F / 4);
  size_t te2 = (size_t)E * (OUT_F / 4);
  agg_self <<<(t2 + B - 1) / B, B, 0, stream>>>(h2, dinv, b2, out, n, OUT_F, 5, 1);
  agg_edges<<<(te2 + B - 1) / B, B, 0, stream>>>(src, dst, dinv, h2, out, E, OUT_F, 5);
}
